// GNNFeatureSelector_39737037423074
// MI455X (gfx1250) — matrix operand order repaired and checked by reading, not compiled, not run on hardware
//
#include <hip/hip_runtime.h>
#include <hip/hip_bf16.h>
#include <hip/hip_fp16.h>

typedef __attribute__((ext_vector_type(16))) _Float16 v16h;
typedef __attribute__((ext_vector_type(8)))  float    v8f;

#define NND 100000
#define NED 800000
#define CH  128

// ---------------- degree / normalization ----------------

__global__ void k_deg_init(float* __restrict__ deg, int n) {
    int i = blockIdx.x * blockDim.x + threadIdx.x;
    if (i < n) deg[i] = 1.0f;   // self-loop weight
}

__global__ void k_deg_edges(const long long* __restrict__ dst,
                            const float* __restrict__ ew,
                            float* __restrict__ deg, int e) {
    int i = blockIdx.x * blockDim.x + threadIdx.x;
    if (i < e) unsafeAtomicAdd(&deg[(int)dst[i]], ew[i]);
}

__global__ void k_dinv(const float* __restrict__ deg,
                       float* __restrict__ dinv,
                       float* __restrict__ nself, int n) {
    int i = blockIdx.x * blockDim.x + threadIdx.x;
    if (i < n) {
        float d = deg[i];
        float r = d > 0.f ? rsqrtf(d) : 0.f;
        dinv[i]  = r;
        nself[i] = r * r;
    }
}

__global__ void k_norm(const long long* __restrict__ src,
                       const long long* __restrict__ dst,
                       const float* __restrict__ ew,
                       const float* __restrict__ dinv,
                       float* __restrict__ norm, int e) {
    int i = blockIdx.x * blockDim.x + threadIdx.x;
    if (i < e) norm[i] = dinv[(int)src[i]] * ew[i] * dinv[(int)dst[i]];
}

// ---------------- pack W into WMMA B-fragment layout (f16) ----------------
// p = (((nt*4 + k0)*32 + lane) << 4) + i  holds  W[k][n]  with
//   k = k0*32 + (lane>=16 ? 8 : 0) + (i<8 ? i : i+8),  n = nt*16 + (lane&15)
__global__ void k_pack_w(const float* __restrict__ W, _Float16* __restrict__ Wp) {
    int p    = blockIdx.x * blockDim.x + threadIdx.x;   // 16384 elements
    int i    = p & 15;
    int lane = (p >> 4) & 31;
    int k0   = (p >> 9) & 3;
    int nt   = p >> 11;
    int k = k0 * 32 + ((lane >> 4) << 3) + (i < 8 ? i : i + 8);
    int n = (nt << 4) + (lane & 15);
    Wp[p] = (_Float16)W[k * CH + n];
}

// ---------------- WMMA GEMM: out[N,128] = f(A)[N,128] @ W[128,128] ----------------
// block = 256 threads = 8 waves; wave w -> columns [16w,16w+16); block -> 32 rows.
template <bool PRE_BIAS_RELU>
__global__ __launch_bounds__(256) void k_gemm(const float* __restrict__ A,
                                              const _Float16* __restrict__ Wp,
                                              const float* __restrict__ pre_bias,
                                              float* __restrict__ out) {
    const int lane = threadIdx.x & 31;
    const int wv   = threadIdx.x >> 5;       // 0..7 -> N tile
    const int l16  = lane & 15;
    const int hi   = lane >> 4;
    const int r0   = blockIdx.x * 32;

    // B fragments for all 4 K-steps (32B contiguous per lane thanks to packing)
    v16h b[4];
#pragma unroll
    for (int k0 = 0; k0 < 4; ++k0)
        b[k0] = *(const v16h*)(Wp + ((((wv * 4 + k0) * 32) + lane) << 4));

    v8f acc[2] = {};

#pragma unroll
    for (int mt = 0; mt < 2; ++mt) {
        const int row = r0 + mt * 16 + l16;
        const float* __restrict__ arow = A + (size_t)row * CH;
#pragma unroll
        for (int k0 = 0; k0 < 4; ++k0) {
            const int kb = k0 * 32 + hi * 8;
            float4 f0 = *(const float4*)(arow + kb);
            float4 f1 = *(const float4*)(arow + kb + 4);
            float4 f2 = *(const float4*)(arow + kb + 16);
            float4 f3 = *(const float4*)(arow + kb + 20);
            float va[16] = {f0.x, f0.y, f0.z, f0.w, f1.x, f1.y, f1.z, f1.w,
                            f2.x, f2.y, f2.z, f2.w, f3.x, f3.y, f3.z, f3.w};
            v16h a;
#pragma unroll
            for (int i = 0; i < 16; ++i) {
                float v = va[i];
                if (PRE_BIAS_RELU) {
                    int col = kb + (i < 8 ? i : i + 8);
                    v = fmaxf(v + pre_bias[col], 0.f);
                }
                a[i] = (_Float16)v;
            }
            acc[mt] = __builtin_amdgcn_wmma_f32_16x16x32_f16(
                false, a, false, b[k0], (short)0, acc[mt], false, false);
        }
    }

#pragma unroll
    for (int mt = 0; mt < 2; ++mt) {
#pragma unroll
        for (int r = 0; r < 8; ++r) {
            int m = r0 + mt * 16 + hi * 8 + r;
            out[(size_t)m * CH + (wv << 4) + l16] = acc[mt][r];
        }
    }
}

// ---------------- aggregation ----------------

__global__ void k_self_init(const float* __restrict__ h,
                            const float* __restrict__ nself,
                            float* __restrict__ agg) {
    int i = blockIdx.x * blockDim.x + threadIdx.x;   // N*128, exact grid
    agg[i] = h[i] * nself[i >> 7];
}

// one wave per edge: coalesced 512B gather of h[src], 4 fp atomics/lane to agg[dst]
__global__ void k_edge_scatter(const long long* __restrict__ src,
                               const long long* __restrict__ dst,
                               const float* __restrict__ norm,
                               const float* __restrict__ h,
                               float* __restrict__ agg) {
    int t    = blockIdx.x * blockDim.x + threadIdx.x;
    int e    = t >> 5;
    int lane = t & 31;
    if (e >= NED) return;
    int   s = (int)src[e];
    int   d = (int)dst[e];
    float w = norm[e];
    float4 hv = ((const float4*)(h + (size_t)s * CH))[lane];
    float* ap = agg + (size_t)d * CH + lane * 4;
    unsafeAtomicAdd(ap + 0, hv.x * w);
    unsafeAtomicAdd(ap + 1, hv.y * w);
    unsafeAtomicAdd(ap + 2, hv.z * w);
    unsafeAtomicAdd(ap + 3, hv.w * w);
}

// ---------------- final FC + sigmoid: one wave per node ----------------

__global__ void k_final(const float* __restrict__ agg,
                        const float* __restrict__ b2,
                        const float* __restrict__ wfc,
                        const float* __restrict__ bfc,
                        float* __restrict__ out) {
    int t    = blockIdx.x * blockDim.x + threadIdx.x;
    int node = t >> 5;
    int lane = t & 31;
    if (node >= NND) return;
    float4 a  = ((const float4*)(agg + (size_t)node * CH))[lane];
    float4 bb = ((const float4*)b2)[lane];
    float4 w  = ((const float4*)wfc)[lane];
    float p = fmaxf(a.x + bb.x, 0.f) * w.x + fmaxf(a.y + bb.y, 0.f) * w.y +
              fmaxf(a.z + bb.z, 0.f) * w.z + fmaxf(a.w + bb.w, 0.f) * w.w;
#pragma unroll
    for (int off = 16; off > 0; off >>= 1) p += __shfl_xor(p, off, 32);
    if (lane == 0) out[node] = 1.f / (1.f + expf(-(p + bfc[0])));
}

// ---------------- launch ----------------

extern "C" void kernel_launch(void* const* d_in, const int* in_sizes, int n_in,
                              void* d_out, int out_size, void* d_ws, size_t ws_size,
                              hipStream_t stream) {
    const float*     x    = (const float*)d_in[0];
    const long long* ei   = (const long long*)d_in[1];   // int64 [2,E]
    const float*     ew   = (const float*)d_in[2];
    const float*     W1   = (const float*)d_in[3];
    const float*     b1   = (const float*)d_in[4];
    const float*     W2   = (const float*)d_in[5];
    const float*     b2   = (const float*)d_in[6];
    const float*     Wfc  = (const float*)d_in[7];
    const float*     bfc  = (const float*)d_in[8];
    const long long* srcI = ei;
    const long long* dstI = ei + NED;
    float*           out  = (float*)d_out;

    // workspace layout (bytes)
    char* ws = (char*)d_ws;
    float*     deg   = (float*)(ws + 0);                       // N   (becomes dinv)
    float*     nself = (float*)(ws + 400000);                  // N
    float*     norm  = (float*)(ws + 800000);                  // E
    float*     t     = (float*)(ws + 4000000);                 // N*128
    float*     agg   = (float*)(ws + 55200000);                // N*128
    _Float16*  W1p   = (_Float16*)(ws + 106400000);            // 16384
    _Float16*  W2p   = (_Float16*)(ws + 106432768);            // 16384

    const int TB = 256;

    // weights -> WMMA fragment layout
    k_pack_w<<<64, TB, 0, stream>>>(W1, W1p);
    k_pack_w<<<64, TB, 0, stream>>>(W2, W2p);

    // GCN normalization
    k_deg_init<<<(NND + TB - 1) / TB, TB, 0, stream>>>(deg, NND);
    k_deg_edges<<<NED / TB, TB, 0, stream>>>(dstI, ew, deg, NED);
    k_dinv<<<(NND + TB - 1) / TB, TB, 0, stream>>>(deg, deg, nself, NND);
    k_norm<<<NED / TB, TB, 0, stream>>>(srcI, dstI, ew, deg, norm, NED);

    // layer 1: t = x @ W1 ; agg = self + scatter
    k_gemm<false><<<NND / 32, TB, 0, stream>>>(x, W1p, nullptr, t);
    k_self_init<<<NND * CH / TB, TB, 0, stream>>>(t, nself, agg);
    k_edge_scatter<<<NED * 32 / TB, TB, 0, stream>>>(srcI, dstI, norm, t, agg);

    // layer 2: t = relu(agg + b1) @ W2 ; agg = self + scatter
    k_gemm<true><<<NND / 32, TB, 0, stream>>>(agg, W2p, b1, t);
    k_self_init<<<NND * CH / TB, TB, 0, stream>>>(t, nself, agg);
    k_edge_scatter<<<NED * 32 / TB, TB, 0, stream>>>(srcI, dstI, norm, t, agg);

    // head: sigmoid(relu(agg + b2) @ Wfc + bfc)
    k_final<<<NND * 32 / TB, TB, 0, stream>>>(agg, b2, Wfc, bfc, out);
}
